// BEVDeformableLayer_11433202942359
// MI455X (gfx1250) — matrix operand order repaired and checked by reading, not compiled, not run on hardware
//
#include <hip/hip_runtime.h>
#include <hip/hip_bf16.h>
#include <math.h>

typedef __attribute__((ext_vector_type(16))) _Float16 v16h;
typedef __attribute__((ext_vector_type(8)))  _Float16 v8h;
typedef __attribute__((ext_vector_type(8)))  float    v8f;
typedef __attribute__((ext_vector_type(4)))  float    v4f;

#define NTOKc 2500
#define TPADc 2528        // padded token count (multiple of 32) for attention tiles
#define MLc   1500
#define BNcc  6
#define Dcc   256
#define DFFc  1024
#define NHcc  8
#define DHcc  32
#define LINcc 14960

static __device__ __forceinline__ v8f vzero8() {
  v8f z;
#pragma unroll
  for (int i = 0; i < 8; ++i) z[i] = 0.f;
  return z;
}

// Async global->LDS copy of 16 bytes per lane (gfx1250, ASYNCcnt-tracked).
// lds_off: wave-relative LDS byte address; goff: 32-bit byte offset vs saddr.
static __device__ __forceinline__ void async_cp16(uint32_t lds_off, uint32_t goff,
                                                  const void* sbase) {
  asm volatile("global_load_async_to_lds_b128 %0, %1, %2"
               :: "v"(lds_off), "v"(goff), "s"((uint64_t)(uintptr_t)sbase)
               : "memory");
}

// ---------------------------------------------------------------------------
// WMMA GEMM with async-LDS double buffering:
//   Y[M,N] = X[M,K] @ W[N,K]^T + bias[N]  (+res, +relu)
// MODE 0: f32 row-major output.  MODE 1: f16 "value" layout [BN][NH][LIN][32].
// Block = 256 threads = 8 waves (4 along M x 2 along N); block tile 64M x 128N.
// Per 32-K chunk: stage X tile (64x32 f32, 8KB) + W tile (128x32 f32, 16KB)
// into LDS via GLOBAL_LOAD_ASYNC_TO_LDS_B128; s_wait_asynccnt + barrier.
// ---------------------------------------------------------------------------
template <int MODE>
__global__ __launch_bounds__(256)
void gemm_wmma(const float* __restrict__ X, const float* __restrict__ W,
               const float* __restrict__ bias, const float* __restrict__ res,
               void* __restrict__ Yv, int M, int N, int K, int relu)
{
  __shared__ __align__(16) float Xs[2][64 * 32];
  __shared__ __align__(16) float Ws[2][128 * 32];

  const int tid  = threadIdx.x;
  const int lane = tid & 31;
  const int wid  = tid >> 5;
  const int wm   = wid & 3;
  const int wn   = wid >> 2;
  const int l16  = lane & 15;
  const int hi   = lane >> 4;                 // half-wave: 0 or 1

  const int mblk  = blockIdx.x * 64;
  const int nblk  = blockIdx.y * 128;
  const int m0    = mblk + wm * 16;
  const int nbase = nblk + wn * 64;

  const int ab0 = hi ? 16 : 0;                // A-operand K base within chunk
  const int bd0 = hi ? 16 : 0;                // B-operand K base within chunk

  // Issue 6 async b128 copies per thread-block chunk: 2 for X, 4 for W.
  auto issue_copy = [&](int buf, int kc) {
#pragma unroll
    for (int i = 0; i < 2; ++i) {             // X: 512 segments of 16B
      const int seg = tid + i * 256;
      const int row = seg >> 3, c4 = (seg & 7) * 4;
      const int grow = min(mblk + row, M - 1);
      const uint32_t goff = (uint32_t)(((size_t)grow * K + kc + c4) * 4);
      const uint32_t loff = (uint32_t)(uintptr_t)&Xs[buf][row * 32 + c4];
      async_cp16(loff, goff, X);
    }
#pragma unroll
    for (int i = 0; i < 4; ++i) {             // W: 1024 segments of 16B
      const int seg = tid + i * 256;
      const int row = seg >> 3, c4 = (seg & 7) * 4;
      const uint32_t goff = (uint32_t)(((size_t)(nblk + row) * K + kc + c4) * 4);
      const uint32_t loff = (uint32_t)(uintptr_t)&Ws[buf][row * 32 + c4];
      async_cp16(loff, goff, W);
    }
  };

  v8f c[4];
#pragma unroll
  for (int t = 0; t < 4; ++t) c[t] = vzero8();

  const int nchunk = K >> 5;
  issue_copy(0, 0);

  for (int ic = 0; ic < nchunk; ++ic) {
    const int buf = ic & 1;
    if (ic + 1 < nchunk) {
      issue_copy(buf ^ 1, (ic + 1) * 32);
      asm volatile("s_wait_asynccnt 6" ::: "memory");  // chunk ic landed
    } else {
      asm volatile("s_wait_asynccnt 0" ::: "memory");
    }
    __syncthreads();                                   // publish all waves' copies

    // ---- A operand from LDS (f32 -> f16)
    v16h a;
    {
      const float* xr = &Xs[buf][(wm * 16 + l16) * 32 + ab0];
#pragma unroll
      for (int j = 0; j < 8; ++j) {
        a[j]     = (_Float16)xr[j];
        a[8 + j] = (_Float16)xr[8 + j];
      }
    }
#pragma unroll
    for (int t = 0; t < 4; ++t) {
      const float* wr = &Ws[buf][(wn * 64 + t * 16 + l16) * 32 + bd0];
      v16h b;
#pragma unroll
      for (int j = 0; j < 16; ++j) b[j] = (_Float16)wr[j];
      c[t] = __builtin_amdgcn_wmma_f32_16x16x32_f16(false, a, false, b,
                                                    (short)0, c[t], false, false);
    }
    __syncthreads();                                   // buffer reuse guard
  }

  // ---- epilogue (C layout: vgpr r, lane -> row m0+r+8*hi, col tile_n0+l16)
#pragma unroll
  for (int t = 0; t < 4; ++t) {
    const int   col = nbase + t * 16 + l16;
    const float bv  = bias[col];
#pragma unroll
    for (int r = 0; r < 8; ++r) {
      const int row = m0 + r + 8 * hi;
      if (row < M) {
        float v = c[t][r] + bv;
        if (MODE == 0) {
          if (res)  v += res[(size_t)row * N + col];
          if (relu) v = fmaxf(v, 0.f);
          ((float*)Yv)[(size_t)row * N + col] = v;
        } else {
          const int bn = row / LINcc, i = row % LINcc;
          const int h  = col >> 5,   d = col & 31;
          ((_Float16*)Yv)[(((size_t)(bn * NHcc + h)) * LINcc + i) * 32 + d] = (_Float16)v;
        }
      }
    }
  }
}

// ---------------------------------------------------------------------------
// q = tgt + query_pos
// ---------------------------------------------------------------------------
__global__ void add_kernel(const float* __restrict__ a, const float* __restrict__ b,
                           float* __restrict__ o, int n)
{
  int i = blockIdx.x * 256 + threadIdx.x;
  if (i < n) o[i] = a[i] + b[i];
}

// ---------------------------------------------------------------------------
// Pack Q,K,V projections (f32 [T,256]) into attention layouts (f16, zero-padded):
//   Qh,Kh: [NH][TPAD][32]   Vt: [NH][32][TPAD]
// ---------------------------------------------------------------------------
__global__ __launch_bounds__(256)
void pack_qkv(const float* __restrict__ Qf, const float* __restrict__ Kf,
              const float* __restrict__ Vf,
              _Float16* __restrict__ Qh, _Float16* __restrict__ Kh,
              _Float16* __restrict__ Vt)
{
  const int t = blockIdx.x;          // 0..TPAD-1
  const int ccol = threadIdx.x;      // 0..255
  const int h = ccol >> 5, d = ccol & 31;
  float q = 0.f, k = 0.f, v = 0.f;
  if (t < NTOKc) {
    q = Qf[(size_t)t * Dcc + ccol];
    k = Kf[(size_t)t * Dcc + ccol];
    v = Vf[(size_t)t * Dcc + ccol];
  }
  Qh[((size_t)h * TPADc + t) * 32 + d] = (_Float16)q;
  Kh[((size_t)h * TPADc + t) * 32 + d] = (_Float16)k;
  Vt[((size_t)h * 32 + d) * TPADc + t] = (_Float16)v;
}

// ---------------------------------------------------------------------------
// Flash attention: one wave = (head, 16-query tile); stream over 32-key chunks.
// S = Q K^T via 2 WMMA; online softmax with half-wave shuffles; P restaged
// through per-wave LDS into A-operand layout; O += P V via 2 WMMA.
// ---------------------------------------------------------------------------
__global__ __launch_bounds__(256)
void attn_kernel(const _Float16* __restrict__ Qh, const _Float16* __restrict__ Kh,
                 const _Float16* __restrict__ Vt, float* __restrict__ sa)
{
  __shared__ float pst[8][16 * 32];
  const int lane = threadIdx.x & 31;
  const int wv   = threadIdx.x >> 5;
  const int g    = blockIdx.x * 8 + wv;      // 0..1255
  const int qt   = g >> 3;                   // 0..156
  const int h    = g & 7;
  const int q0   = qt * 16;
  const int l16  = lane & 15, hi = lane >> 4;

  // Q A-operand (rows q0..q0+15, padded region is zero)
  v16h aq;
  {
    const _Float16* qrow = Qh + ((size_t)h * TPADc + (q0 + l16)) * 32 + (hi ? 16 : 0);
    v8h a0 = *(const v8h*)(qrow);
    v8h a1 = *(const v8h*)(qrow + 8);
#pragma unroll
    for (int j = 0; j < 8; ++j) { aq[j] = a0[j]; aq[8 + j] = a1[j]; }
  }

  v8f o0 = vzero8(), o1 = vzero8(), li = vzero8(), mi;
#pragma unroll
  for (int r = 0; r < 8; ++r) mi[r] = -1e30f;

  const float scale = 0.17677669529663689f;   // 1/sqrt(32)
  float* myp = pst[wv];

  for (int kb = 0; kb < NTOKc; kb += 32) {
    // K B-operands: two 16-key tiles, 16 consecutive dh per lane
    const _Float16* kp0 = Kh + ((size_t)h * TPADc + (kb + l16)) * 32 + (hi ? 16 : 0);
    const _Float16* kp1 = Kh + ((size_t)h * TPADc + (kb + 16 + l16)) * 32 + (hi ? 16 : 0);
    v16h bk0 = *(const v16h*)kp0;
    v16h bk1 = *(const v16h*)kp1;

    v8f z = vzero8();
    v8f s0 = __builtin_amdgcn_wmma_f32_16x16x32_f16(false, aq, false, bk0, (short)0, z, false, false);
    v8f s1 = __builtin_amdgcn_wmma_f32_16x16x32_f16(false, aq, false, bk1, (short)0, z, false, false);

    const bool v0 = (kb + l16) < NTOKc;
    const bool v1 = (kb + 16 + l16) < NTOKc;
#pragma unroll
    for (int r = 0; r < 8; ++r) {
      s0[r] = v0 ? s0[r] * scale : -1e30f;
      s1[r] = v1 ? s1[r] * scale : -1e30f;
    }

    v8f p0, p1;
#pragma unroll
    for (int r = 0; r < 8; ++r) {
      float mx = fmaxf(s0[r], s1[r]);
      mx = fmaxf(mx, __shfl_xor(mx, 1, 32));
      mx = fmaxf(mx, __shfl_xor(mx, 2, 32));
      mx = fmaxf(mx, __shfl_xor(mx, 4, 32));
      mx = fmaxf(mx, __shfl_xor(mx, 8, 32));
      const float mn   = fmaxf(mi[r], mx);
      const float corr = __expf(mi[r] - mn);
      mi[r] = mn;
      p0[r] = __expf(s0[r] - mn);
      p1[r] = __expf(s1[r] - mn);
      float rs = p0[r] + p1[r];
      rs += __shfl_xor(rs, 1, 32);
      rs += __shfl_xor(rs, 2, 32);
      rs += __shfl_xor(rs, 4, 32);
      rs += __shfl_xor(rs, 8, 32);
      li[r] = li[r] * corr + rs;
      o0[r] *= corr;
      o1[r] *= corr;
    }

    // stage P (C layout) -> LDS -> reload in A-operand layout
#pragma unroll
    for (int r = 0; r < 8; ++r) {
      myp[(r + 8 * hi) * 32 + l16]      = p0[r];
      myp[(r + 8 * hi) * 32 + 16 + l16] = p1[r];
    }
    v16h ap;
    {
      const float* pr = myp + l16 * 32 + (hi ? 16 : 0);
#pragma unroll
      for (int j = 0; j < 8; ++j) {
        ap[j]     = (_Float16)pr[j];
        ap[8 + j] = (_Float16)pr[8 + j];
      }
    }

    // V B-operands from transposed layout (16 consecutive keys per lane)
    const _Float16* vb = Vt + (size_t)h * 32 * TPADc;
    v16h bv0 = *(const v16h*)(vb + (size_t)(l16) * TPADc + kb + (hi ? 16 : 0));
    v16h bv1 = *(const v16h*)(vb + (size_t)(16 + l16) * TPADc + kb + (hi ? 16 : 0));
    o0 = __builtin_amdgcn_wmma_f32_16x16x32_f16(false, ap, false, bv0, (short)0, o0, false, false);
    o1 = __builtin_amdgcn_wmma_f32_16x16x32_f16(false, ap, false, bv1, (short)0, o1, false, false);
  }

#pragma unroll
  for (int r = 0; r < 8; ++r) {
    const int row = q0 + r + 8 * hi;
    if (row < NTOKc) {
      const float inv = 1.f / li[r];
      sa[(size_t)row * Dcc + h * 32 + l16]      = o0[r] * inv;
      sa[(size_t)row * Dcc + h * 32 + 16 + l16] = o1[r] * inv;
    }
  }
}

// ---------------------------------------------------------------------------
// Row LayerNorm over D=256: one wave per row.
// ---------------------------------------------------------------------------
__global__ __launch_bounds__(256)
void ln_kernel(const float* __restrict__ in, const float* __restrict__ g,
               const float* __restrict__ b, float* __restrict__ out, int rows)
{
  const int lane = threadIdx.x & 31, wv = threadIdx.x >> 5;
  const int r = blockIdx.x * 8 + wv;
  if (r >= rows) return;
  const float* x = in + (size_t)r * Dcc;
  float v[8], s = 0.f, s2 = 0.f;
#pragma unroll
  for (int j = 0; j < 8; ++j) {
    v[j] = x[lane + 32 * j];
    s += v[j]; s2 += v[j] * v[j];
  }
#pragma unroll
  for (int o = 16; o > 0; o >>= 1) {
    s  += __shfl_xor(s,  o, 32);
    s2 += __shfl_xor(s2, o, 32);
  }
  const float mean = s * (1.f / Dcc);
  const float var  = s2 * (1.f / Dcc) - mean * mean;
  const float rstd = rsqrtf(var + 1e-5f);
  float* y = out + (size_t)r * Dcc;
#pragma unroll
  for (int j = 0; j < 8; ++j) {
    const int col = lane + 32 * j;
    y[col] = (v[j] - mean) * rstd * g[col] + b[col];
  }
}

// ---------------------------------------------------------------------------
// Per-camera gather: tgt_rb, qd = (x+qpos)[idx]*mask, ref_rb
// ---------------------------------------------------------------------------
__global__ __launch_bounds__(256)
void gather_kernel(const int* __restrict__ cam_idx, const int* __restrict__ cam_len,
                   const float* __restrict__ x, const float* __restrict__ qpos,
                   const float* __restrict__ refp,
                   float* __restrict__ qd, float* __restrict__ tgt_rb,
                   float* __restrict__ ref_rb)
{
  const int row = blockIdx.x;             // 0..8999
  const int bn = row / MLc, qq = row % MLc;
  const int idx = cam_idx[bn * MLc + qq];
  const bool m = qq < cam_len[bn];
  const int t = threadIdx.x;
  const float xv = m ? x[(size_t)idx * Dcc + t] : 0.f;
  const float qp = m ? qpos[(size_t)idx * Dcc + t] : 0.f;
  tgt_rb[(size_t)row * Dcc + t] = xv;
  qd[(size_t)row * Dcc + t]     = xv + qp;
  if (t < 8) ref_rb[row * 8 + t] = m ? refp[((size_t)bn * NTOKc + idx) * 8 + t] : 0.f;
}

// ---------------------------------------------------------------------------
// Attention-weight softmax: groups of 32 (one wave each), in place.
// ---------------------------------------------------------------------------
__global__ __launch_bounds__(256)
void softmax_aw(float* __restrict__ aw)
{
  const int lane = threadIdx.x & 31, wv = threadIdx.x >> 5;
  const int gidx = blockIdx.x * 8 + wv;     // 0..71999
  const int row = gidx >> 3, h = gidx & 7;
  const size_t base = (size_t)row * Dcc + h * 32 + lane;
  float v = aw[base];
  float mx = v;
#pragma unroll
  for (int o = 16; o > 0; o >>= 1) mx = fmaxf(mx, __shfl_xor(mx, o, 32));
  const float e = __expf(v - mx);
  float s = e;
#pragma unroll
  for (int o = 16; o > 0; o >>= 1) s += __shfl_xor(s, o, 32);
  aw[base] = e / s;
}

// ---------------------------------------------------------------------------
// Deformable bilinear sampling: one wave per (bn, query, head); lane = channel.
// value: f16 [BN][NH][LIN][32] -> 64B coalesced row gathers.
// ---------------------------------------------------------------------------
__global__ __launch_bounds__(256)
void sample_kernel(const float* __restrict__ ref_rb, const float* __restrict__ off,
                   const float* __restrict__ aw, const _Float16* __restrict__ value,
                   float* __restrict__ acc)
{
  const int Wi[4]  = {176, 88, 44, 22};
  const int Hi[4]  = {64, 32, 16, 8};
  const int LST[4] = {0, 11264, 14080, 14784};

  const int lane = threadIdx.x & 31, wv = threadIdx.x >> 5;
  const int g = blockIdx.x * 8 + wv;        // 0..71999
  const int bn = g / (MLc * NHcc);
  const int rem = g - bn * (MLc * NHcc);
  const int qq = rem >> 3, h = rem & 7;
  const int row = bn * MLc + qq;

  const float* offr = off + (size_t)row * 512 + h * 64;   // [l][s][p][2]
  const float* awr  = aw  + (size_t)row * Dcc + h * 32;   // [l][s][p]
  float a = 0.f;

#pragma unroll
  for (int l = 0; l < 4; ++l) {
    const int Wc = Wi[l], Hc = Hi[l];
    const float Wf = (float)Wc, Hf = (float)Hc;
    const _Float16* vb = value + (((size_t)(bn * NHcc + h)) * LINcc + LST[l]) * 32 + lane;
#pragma unroll
    for (int s = 0; s < 4; ++s) {
      const float rx = ref_rb[row * 8 + s * 2 + 0];
      const float ry = ref_rb[row * 8 + s * 2 + 1];
#pragma unroll
      for (int p = 0; p < 2; ++p) {
        const float ox = offr[((l * 4 + s) * 2 + p) * 2 + 0];
        const float oy = offr[((l * 4 + s) * 2 + p) * 2 + 1];
        const float lx = rx + ox / Wf;
        const float ly = ry + oy / Hf;
        const float xx = lx * Wf - 0.5f, yy = ly * Hf - 0.5f;
        const float x0 = floorf(xx), y0 = floorf(yy);
        const float fx = xx - x0, fy = yy - y0;
        const int ix0 = (int)x0, iy0 = (int)y0;
        float samp = 0.f;
#pragma unroll
        for (int dy = 0; dy < 2; ++dy) {
#pragma unroll
          for (int dx = 0; dx < 2; ++dx) {
            const int xi = ix0 + dx, yi = iy0 + dy;
            const float w = (dx ? fx : 1.f - fx) * (dy ? fy : 1.f - fy);
            const bool valid = (xi >= 0) && (xi < Wc) && (yi >= 0) && (yi < Hc);
            const int xc = min(max(xi, 0), Wc - 1);
            const int yc = min(max(yi, 0), Hc - 1);
            const float gv = (float)vb[(size_t)(yc * Wc + xc) * 32];
            samp += gv * (valid ? w : 0.f);
          }
        }
        a += samp * awr[l * 8 + s * 2 + p];
      }
    }
  }
  acc[(size_t)row * Dcc + h * 32 + lane] = a;
}

// ---------------------------------------------------------------------------
__global__ void zero_kernel(float* __restrict__ p, int n)
{
  int i = blockIdx.x * 256 + threadIdx.x;
  if (i < n) p[i] = 0.f;
}

__global__ __launch_bounds__(256)
void scatter_kernel(const int* __restrict__ cam_idx, const int* __restrict__ cam_len,
                    const float* __restrict__ rbn, float* __restrict__ fin,
                    float* __restrict__ cnt)
{
  const int row = blockIdx.x;
  const int bn = row / MLc, qq = row % MLc;
  if (qq >= cam_len[bn]) return;
  const int idx = cam_idx[bn * MLc + qq];
  const int t = threadIdx.x;
  atomicAdd(&fin[(size_t)idx * Dcc + t], rbn[(size_t)row * Dcc + t]);
  if (t == 0) atomicAdd(&cnt[idx], 1.f);
}

__global__ void finalize_kernel(float* __restrict__ fin, const float* __restrict__ cnt)
{
  const int row = blockIdx.x, t = threadIdx.x;
  fin[(size_t)row * Dcc + t] *= 1.f / (cnt[row] + 0.001f);
}

// ---------------------------------------------------------------------------
extern "C" void kernel_launch(void* const* d_in, const int* in_sizes, int n_in,
                              void* d_out, int out_size, void* d_ws, size_t ws_size,
                              hipStream_t stream)
{
  (void)in_sizes; (void)n_in; (void)out_size; (void)ws_size;

  const int*   cam_indexes = (const int*)d_in[0];
  const int*   cam_lens    = (const int*)d_in[1];
  const float* tgt         = (const float*)d_in[2];
  const float* query_pos   = (const float*)d_in[3];
  const float* refp        = (const float*)d_in[4];
  const float* src         = (const float*)d_in[5];
  const float* in_proj_w   = (const float*)d_in[8];
  const float* in_proj_b   = (const float*)d_in[9];
  const float* mha_out_w   = (const float*)d_in[10];
  const float* mha_out_b   = (const float*)d_in[11];
  const float* ln1_g = (const float*)d_in[12];
  const float* ln1_b = (const float*)d_in[13];
  const float* ln2_g = (const float*)d_in[14];
  const float* ln2_b = (const float*)d_in[15];
  const float* ln3_g = (const float*)d_in[16];
  const float* ln3_b = (const float*)d_in[17];
  const float* off_w = (const float*)d_in[18];
  const float* off_b = (const float*)d_in[19];
  const float* aw_w  = (const float*)d_in[20];
  const float* aw_b  = (const float*)d_in[21];
  const float* val_w = (const float*)d_in[22];
  const float* val_b = (const float*)d_in[23];
  const float* dout_w = (const float*)d_in[24];
  const float* dout_b = (const float*)d_in[25];
  const float* ffn_w1 = (const float*)d_in[26];
  const float* ffn_b1 = (const float*)d_in[27];
  const float* ffn_w2 = (const float*)d_in[28];
  const float* ffn_b2 = (const float*)d_in[29];

  // ---- workspace bump allocator (256B aligned slots)
  char* ws = (char*)d_ws;
  size_t wo = 0;
  auto alloc = [&](size_t bytes) -> char* {
    char* p = ws + wo;
    wo += (bytes + 255) & ~(size_t)255;
    return p;
  };
  const size_t TD = (size_t)NTOKc * Dcc;        // 640000
  const size_t RD = (size_t)BNcc * MLc * Dcc;   // 2304000
  float*    q      = (float*)alloc(TD * 4);
  float*    Qf     = (float*)alloc(TD * 4);
  float*    Kf     = (float*)alloc(TD * 4);
  float*    Vf     = (float*)alloc(TD * 4);
  _Float16* Qh     = (_Float16*)alloc((size_t)NHcc * TPADc * 32 * 2);
  _Float16* Kh     = (_Float16*)alloc((size_t)NHcc * TPADc * 32 * 2);
  _Float16* Vt     = (_Float16*)alloc((size_t)NHcc * 32 * TPADc * 2);
  float*    sa     = (float*)alloc(TD * 4);
  float*    y2     = (float*)alloc(TD * 4);
  float*    x      = (float*)alloc(TD * 4);
  float*    qd     = (float*)alloc(RD * 4);
  float*    tgt_rb = (float*)alloc(RD * 4);
  float*    ref_rb = (float*)alloc((size_t)BNcc * MLc * 8 * 4);
  float*    offb   = (float*)alloc((size_t)BNcc * MLc * 512 * 4);
  float*    awb    = (float*)alloc(RD * 4);
  _Float16* value  = (_Float16*)alloc((size_t)BNcc * NHcc * LINcc * 32 * 2);
  float*    accb   = (float*)alloc(RD * 4);
  float*    y3     = (float*)alloc(RD * 4);
  float*    rbn    = (float*)alloc(RD * 4);
  float*    fin    = (float*)alloc(TD * 4);
  float*    cnt    = (float*)alloc((size_t)NTOKc * 4);
  float*    h1     = (float*)alloc((size_t)NTOKc * DFFc * 4);
  float*    y4     = (float*)alloc(TD * 4);

  const int nT = (int)TD;
  const int gemm_gx_t = (NTOKc + 63) / 64;            // 40
  const int gemm_gx_r = (BNcc * MLc + 63) / 64;       // 141
  const int gemm_gx_v = (BNcc * LINcc + 63) / 64;     // 1403

  // 1. q = tgt + query_pos
  add_kernel<<<(nT + 255) / 256, 256, 0, stream>>>(tgt, query_pos, q, nT);

  // 2. QKV projections (WMMA GEMMs, async-LDS staged)
  gemm_wmma<0><<<dim3(gemm_gx_t, 2), 256, 0, stream>>>(q,   in_proj_w,               in_proj_b,       nullptr, Qf, NTOKc, Dcc, Dcc, 0);
  gemm_wmma<0><<<dim3(gemm_gx_t, 2), 256, 0, stream>>>(q,   in_proj_w + 256 * 256,   in_proj_b + 256, nullptr, Kf, NTOKc, Dcc, Dcc, 0);
  gemm_wmma<0><<<dim3(gemm_gx_t, 2), 256, 0, stream>>>(tgt, in_proj_w + 512 * 256,   in_proj_b + 512, nullptr, Vf, NTOKc, Dcc, Dcc, 0);

  // 3. Pack to f16 attention layouts (zero-padded to TPAD)
  pack_qkv<<<TPADc, 256, 0, stream>>>(Qf, Kf, Vf, Qh, Kh, Vt);

  // 4. Flash attention (WMMA) -> sa
  attn_kernel<<<157, 256, 0, stream>>>(Qh, Kh, Vt, sa);

  // 5. Output projection + residual(tgt), then LN2 -> x
  gemm_wmma<0><<<dim3(gemm_gx_t, 2), 256, 0, stream>>>(sa, mha_out_w, mha_out_b, tgt, y2, NTOKc, Dcc, Dcc, 0);
  ln_kernel<<<(NTOKc + 7) / 8, 256, 0, stream>>>(y2, ln2_g, ln2_b, x, NTOKc);

  // 6. Per-camera gather -> qd, tgt_rb, ref_rb
  gather_kernel<<<BNcc * MLc, 256, 0, stream>>>(cam_indexes, cam_lens, x, query_pos, refp,
                                                qd, tgt_rb, ref_rb);

  // 7. Offset & attention-weight projections
  gemm_wmma<0><<<dim3(gemm_gx_r, 4), 256, 0, stream>>>(qd, off_w, off_b, nullptr, offb, BNcc * MLc, 512, Dcc, 0);
  gemm_wmma<0><<<dim3(gemm_gx_r, 2), 256, 0, stream>>>(qd, aw_w,  aw_b,  nullptr, awb,  BNcc * MLc, Dcc, Dcc, 0);
  softmax_aw<<<BNcc * MLc, 256, 0, stream>>>(awb);

  // 8. Value projection, fused f16 [BN][NH][LIN][32] writeback
  gemm_wmma<1><<<dim3(gemm_gx_v, 2), 256, 0, stream>>>(src, val_w, val_b, nullptr, value,
                                                       BNcc * LINcc, Dcc, Dcc, 0);

  // 9. Deformable bilinear sampling -> acc
  sample_kernel<<<BNcc * MLc, 256, 0, stream>>>(ref_rb, offb, awb, value, accb);

  // 10. Deform output projection + residual(tgt_rb), then LN1 -> rbn
  gemm_wmma<0><<<dim3(gemm_gx_r, 2), 256, 0, stream>>>(accb, dout_w, dout_b, tgt_rb, y3, BNcc * MLc, Dcc, Dcc, 0);
  ln_kernel<<<(BNcc * MLc + 7) / 8, 256, 0, stream>>>(y3, ln1_g, ln1_b, rbn, BNcc * MLc);

  // 11. Scatter-average back to tokens
  zero_kernel<<<(nT + 255) / 256, 256, 0, stream>>>(fin, nT);
  zero_kernel<<<(NTOKc + 255) / 256, 256, 0, stream>>>(cnt, NTOKc);
  scatter_kernel<<<BNcc * MLc, 256, 0, stream>>>(cam_indexes, cam_lens, rbn, fin, cnt);
  finalize_kernel<<<NTOKc, 256, 0, stream>>>(fin, cnt);

  // 12. FFN: relu(final @ w1^T + b1) @ w2^T + b2 + final, then LN3 -> out
  gemm_wmma<0><<<dim3(gemm_gx_t, 8), 256, 0, stream>>>(fin, ffn_w1, ffn_b1, nullptr, h1, NTOKc, DFFc, Dcc, 1);
  gemm_wmma<0><<<dim3(gemm_gx_t, 2), 256, 0, stream>>>(h1, ffn_w2, ffn_b2, fin, y4, NTOKc, Dcc, DFFc, 0);
  ln_kernel<<<(NTOKc + 7) / 8, 256, 0, stream>>>(y4, ln3_g, ln3_b, (float*)d_out, NTOKc);
}
